// TempAttNet_25993142075624
// MI455X (gfx1250) — matrix operand order repaired and checked by reading, not compiled, not run on hardware
//
#include <hip/hip_runtime.h>
#include <hip/hip_bf16.h>

#define T_STEPS 20
#define NFEAT   256
#define NHID    128
#define NNODE   1024
#define ALPHA_LRELU 0.2f
#define NEG_BIG (-9000000000000000.0f)

typedef __attribute__((ext_vector_type(16))) _Float16 v16h;
typedef __attribute__((ext_vector_type(8)))  _Float16 v8h;
typedef __attribute__((ext_vector_type(8)))  float    v8f;

static __device__ __forceinline__ v8h cvt8(float4 a, float4 b) {
  v8h r;
  r[0] = (_Float16)a.x; r[1] = (_Float16)a.y; r[2] = (_Float16)a.z; r[3] = (_Float16)a.w;
  r[4] = (_Float16)b.x; r[5] = (_Float16)b.y; r[6] = (_Float16)b.z; r[7] = (_Float16)b.w;
  return r;
}
static __device__ __forceinline__ v16h cat16(v8h lo, v8h hi) {
  return __builtin_shufflevector(lo, hi, 0,1,2,3,4,5,6,7,8,9,10,11,12,13,14,15);
}
static __device__ __forceinline__ v8f wmma_f16(v16h a, v16h b, v8f c) {
  // (neg_a, A, neg_b, B, c_mod, C, reuse_a, reuse_b)
  return __builtin_amdgcn_wmma_f32_16x16x32_f16(false, a, false, b, (short)0, c, false, false);
}
static __device__ __forceinline__ float eluf(float v) {
  return v > 0.f ? v : (__expf(v) - 1.f);
}

// ---------------------------------------------------------------------------
// Kernel 1: x = ELU(X @ Win^T + b); Wh = x @ W_share;
//           store WhT (f16, [t][h][node]) + s1 = Wh.a1, s2 = Wh.a2
// Block = 256 threads (8 waves); each wave owns 16 rows of M = T*N = 20480.
// ---------------------------------------------------------------------------
__global__ __launch_bounds__(256) void tgat_proj_kernel(
    const float* __restrict__ node_feat,   // (N, T, F)
    const float* __restrict__ W_share,     // (H, H)
    const float* __restrict__ Win_w,       // (H, F)
    const float* __restrict__ Win_b,       // (H)
    const float* __restrict__ a_att,       // (2H)
    _Float16* __restrict__ whT,            // ws: (T, H, N) f16
    float* __restrict__ s1g,               // ws: (T, N)
    float* __restrict__ s2g)               // ws: (T, N)
{
  // 64 KB LDS, re-used across two phases:
  //   phase 1: whole array = Win_w as f16 (B-layout friendly: [h][f], 32768 halves)
  //   phase 2: [0,16384)  = W_share transposed f16  [k'][h]
  //            [16384, )  = per-wave x tiles (16x128 f16, A-layout friendly)
  __shared__ __attribute__((aligned(32))) _Float16 smem[NHID * NFEAT];

  const int tid = threadIdx.x;
  const int l   = tid & 31;
  const int w   = tid >> 5;
  const int ln  = l & 15;
  const int g   = l >> 4;      // half-wave
  const int kbA = g * 16;      // A operand K-base (contiguous 16-run)
  const int kbB = g * 16;      // B operand K-base (contiguous 16-run)

  // ---- phase 1: stage Win_w (f16) -----------------------------------------
  for (int i = tid; i < NHID * NFEAT; i += 256) smem[i] = (_Float16)Win_w[i];
  __syncthreads();

  const int gm0      = blockIdx.x * 128 + w * 16;  // tile base row in (t*N + n)
  const int t        = gm0 >> 10;
  const int nodeBase = gm0 & 1023;
  const float* arow  = node_feat + ((size_t)(nodeBase + ln) * T_STEPS + t) * NFEAT;

  v8f zero = {};
  v8f acc[8];
#pragma unroll
  for (int nt = 0; nt < 8; ++nt) acc[nt] = zero;

  // ---- GEMM1: x = X @ Win^T  (K = 256, 8 k-steps, 64 WMMAs) ---------------
#pragma unroll
  for (int ks = 0; ks < 8; ++ks) {
    const int k0 = ks * 32;
    const float* ap = arow + k0 + kbA;
    v16h a = cat16(cvt8(*(const float4*)ap,        *(const float4*)(ap + 4)),
                   cvt8(*(const float4*)(ap + 8), *(const float4*)(ap + 12)));
#pragma unroll
    for (int nt = 0; nt < 8; ++nt) {
      v16h b = *(const v16h*)(smem + (nt * 16 + ln) * NFEAT + k0 + kbB);
      acc[nt] = wmma_f16(a, b, acc[nt]);
    }
  }
  __syncthreads();  // everyone done reading Win_w staging

  // ---- phase 2 staging: W_share^T (f16) + x tile (bias + ELU, f16) --------
  _Float16* bsh2 = smem;                    // [k'][h] : 16384 halves
  _Float16* xs_w = smem + NHID * NHID + w * (16 * NHID);
  for (int i = tid; i < NHID * NHID; i += 256) {
    int h = i >> 7, kp = i & 127;
    bsh2[kp * NHID + h] = (_Float16)W_share[i];
  }
#pragma unroll
  for (int nt = 0; nt < 8; ++nt) {
    float bb = Win_b[nt * 16 + ln];
#pragma unroll
    for (int r = 0; r < 8; ++r) {
      float v = eluf(acc[nt][r] + bb);
      xs_w[(r + 8 * g) * NHID + nt * 16 + ln] = (_Float16)v;  // [m][h]
    }
  }
  __syncthreads();

  // ---- GEMM2: Wh = x @ W_share  (K = 128, 4 k-steps, 32 WMMAs) ------------
#pragma unroll
  for (int nt = 0; nt < 8; ++nt) acc[nt] = zero;
#pragma unroll
  for (int ks = 0; ks < 4; ++ks) {
    const int kc = ks * 32;
    v16h a = cat16(*(const v8h*)(xs_w + ln * NHID + kc + kbA),
                   *(const v8h*)(xs_w + ln * NHID + kc + kbA + 8));
#pragma unroll
    for (int nt = 0; nt < 8; ++nt) {
      v16h b = *(const v16h*)(bsh2 + (nt * 16 + ln) * NHID + kc + kbB);
      acc[nt] = wmma_f16(a, b, acc[nt]);
    }
  }

  // ---- store WhT f16: whT[(t*H + h)*N + node] (16B per lane per nt) -------
#pragma unroll
  for (int nt = 0; nt < 8; ++nt) {
    v8h pk;
#pragma unroll
    for (int r = 0; r < 8; ++r) pk[r] = (_Float16)acc[nt][r];
    _Float16* dst = whT + ((size_t)t * NHID + nt * 16 + ln) * NNODE + nodeBase + 8 * g;
    *(v8h*)dst = pk;
  }

  // ---- attention scalars: s1 = Wh.a1, s2 = Wh.a2 (16-lane reductions) -----
  float a1v[8], a2v[8];
#pragma unroll
  for (int nt = 0; nt < 8; ++nt) {
    a1v[nt] = a_att[nt * 16 + ln];
    a2v[nt] = a_att[NHID + nt * 16 + ln];
  }
#pragma unroll
  for (int r = 0; r < 8; ++r) {
    float p1 = 0.f, p2 = 0.f;
#pragma unroll
    for (int nt = 0; nt < 8; ++nt) {
      float v = acc[nt][r];
      p1 = fmaf(v, a1v[nt], p1);
      p2 = fmaf(v, a2v[nt], p2);
    }
#pragma unroll
    for (int off = 1; off < 16; off <<= 1) {
      p1 += __shfl_xor(p1, off, 32);
      p2 += __shfl_xor(p2, off, 32);
    }
    if (ln == 0) {
      int node = nodeBase + 8 * g + r;
      s1g[t * NNODE + node] = p1;
      s2g[t * NNODE + node] = p2;
    }
  }
}

// ---------------------------------------------------------------------------
// Kernel 2: masked-softmax attention + aggregation
//   e[i,j] = lrelu(s1_i + s2_j) (masked), att = softmax_j, written to out;
//   h' = att @ Wh (WMMA), feats = ELU(h') written to out.
// Block = 256 threads; wave owns 16 rows i; grid = 20 t * 8 row-blocks.
// ---------------------------------------------------------------------------
static __device__ __forceinline__ float masked_e(float s1l, float u, int q) {
  float e = s1l + u;
  e = e > 0.f ? e : ALPHA_LRELU * e;
  return q > 0 ? e : NEG_BIG;
}

// compute 8 consecutive att values for one row, store them, return f16 pack
static __device__ __forceinline__ v8h att_run8(
    const int* __restrict__ adjrow, const float* __restrict__ s2sh,
    float* __restrict__ attrow, int j, float s1l, float m, float invs)
{
  int4   q0 = *(const int4*)(adjrow + j);
  int4   q1 = *(const int4*)(adjrow + j + 4);
  float4 u0 = *(const float4*)(s2sh + j);
  float4 u1 = *(const float4*)(s2sh + j + 4);
  float f0 = __expf(masked_e(s1l, u0.x, q0.x) - m) * invs;
  float f1 = __expf(masked_e(s1l, u0.y, q0.y) - m) * invs;
  float f2 = __expf(masked_e(s1l, u0.z, q0.z) - m) * invs;
  float f3 = __expf(masked_e(s1l, u0.w, q0.w) - m) * invs;
  float f4 = __expf(masked_e(s1l, u1.x, q1.x) - m) * invs;
  float f5 = __expf(masked_e(s1l, u1.y, q1.y) - m) * invs;
  float f6 = __expf(masked_e(s1l, u1.z, q1.z) - m) * invs;
  float f7 = __expf(masked_e(s1l, u1.w, q1.w) - m) * invs;
  *(float4*)(attrow + j)     = make_float4(f0, f1, f2, f3);
  *(float4*)(attrow + j + 4) = make_float4(f4, f5, f6, f7);
  v8h r;
  r[0] = (_Float16)f0; r[1] = (_Float16)f1; r[2] = (_Float16)f2; r[3] = (_Float16)f3;
  r[4] = (_Float16)f4; r[5] = (_Float16)f5; r[6] = (_Float16)f6; r[7] = (_Float16)f7;
  return r;
}

__global__ __launch_bounds__(256) void tgat_att_kernel(
    const int* __restrict__ adj,          // (T, N, N)
    const _Float16* __restrict__ whT,     // (T, H, N) f16
    const float* __restrict__ s1g,        // (T, N)
    const float* __restrict__ s2g,        // (T, N)
    float* __restrict__ out)              // feats (T*N, H) ++ att (T, N, N)
{
  __shared__ __attribute__((aligned(16))) float s2_sh[NNODE];
  const int tid = threadIdx.x;
  const int t   = blockIdx.x >> 3;
  const int ib  = blockIdx.x & 7;
  for (int i = tid; i < NNODE; i += 256) s2_sh[i] = s2g[t * NNODE + i];
  __syncthreads();

  const int l   = tid & 31;
  const int w   = tid >> 5;
  const int ln  = l & 15;
  const int g   = l >> 4;
  const int kbA = 16 * g;
  const int kbB = 16 * g;
  const int it0 = ib * 128 + w * 16;
  const int row = it0 + ln;
  const float s1l = s1g[t * NNODE + row];
  const int* adjrow = adj + ((size_t)t << 20) + ((size_t)row << 10);

  // ---- pass A1: row max (each lane-pair of a row splits j space) ----------
  float m = NEG_BIG;
  for (int it = 0; it < 128; ++it) {
    int j = ((it << 1) | g) << 2;
    int4   q = *(const int4*)(adjrow + j);
    float4 u = *(const float4*)(s2_sh + j);
    m = fmaxf(m, masked_e(s1l, u.x, q.x));
    m = fmaxf(m, masked_e(s1l, u.y, q.y));
    m = fmaxf(m, masked_e(s1l, u.z, q.z));
    m = fmaxf(m, masked_e(s1l, u.w, q.w));
  }
  m = fmaxf(m, __shfl_xor(m, 16, 32));

  // ---- pass A2: sum of exp ------------------------------------------------
  float s = 0.f;
  for (int it = 0; it < 128; ++it) {
    int j = ((it << 1) | g) << 2;
    int4   q = *(const int4*)(adjrow + j);
    float4 u = *(const float4*)(s2_sh + j);
    s += __expf(masked_e(s1l, u.x, q.x) - m);
    s += __expf(masked_e(s1l, u.y, q.y) - m);
    s += __expf(masked_e(s1l, u.z, q.z) - m);
    s += __expf(masked_e(s1l, u.w, q.w) - m);
  }
  s += __shfl_xor(s, 16, 32);
  const float invs = 1.f / s;

  // ---- pass B: att (write-once) + h' = att @ Wh via WMMA ------------------
  v8f zero = {};
  v8f acc[8];
#pragma unroll
  for (int nt = 0; nt < 8; ++nt) acc[nt] = zero;

  float* attrow = out + (size_t)T_STEPS * NNODE * NHID
                      + ((size_t)t << 20) + ((size_t)row << 10);
  const _Float16* bbase = whT + (size_t)t * NHID * NNODE;

  for (int jc = 0; jc < NNODE; jc += 32) {
    int j1 = jc + kbA;
    v8h alo = att_run8(adjrow, s2_sh, attrow, j1,      s1l, m, invs);
    v8h ahi = att_run8(adjrow, s2_sh, attrow, j1 + 8, s1l, m, invs);
    v16h a = cat16(alo, ahi);
#pragma unroll
    for (int nt = 0; nt < 8; ++nt) {
      const _Float16* bp = bbase + ((size_t)(nt * 16 + ln) << 10) + jc + kbB;
      v16h b = *(const v16h*)bp;           // contiguous 32B per lane
      acc[nt] = wmma_f16(a, b, acc[nt]);
    }
  }

  // ---- epilogue: feats = ELU(h') ------------------------------------------
#pragma unroll
  for (int nt = 0; nt < 8; ++nt) {
#pragma unroll
    for (int r = 0; r < 8; ++r) {
      float v = eluf(acc[nt][r]);
      int irow = it0 + 8 * g + r;
      out[((size_t)(t * NNODE + irow)) * NHID + nt * 16 + ln] = v;
    }
  }
}

// ---------------------------------------------------------------------------
extern "C" void kernel_launch(void* const* d_in, const int* in_sizes, int n_in,
                              void* d_out, int out_size, void* d_ws, size_t ws_size,
                              hipStream_t stream) {
  const float* node_feat = (const float*)d_in[0];
  const int*   adj       = (const int*)d_in[1];
  const float* W_share   = (const float*)d_in[2];
  const float* Win_w     = (const float*)d_in[3];
  const float* Win_b     = (const float*)d_in[4];
  const float* a_att     = (const float*)d_in[5];
  float* out = (float*)d_out;

  _Float16* whT = (_Float16*)d_ws;   // 20*128*1024 f16 = 5,242,880 B
  float* s1g = (float*)((char*)d_ws + (size_t)T_STEPS * NHID * NNODE * sizeof(_Float16));
  float* s2g = s1g + T_STEPS * NNODE;

  tgat_proj_kernel<<<dim3(160), dim3(256), 0, stream>>>(
      node_feat, W_share, Win_w, Win_b, a_att, whT, s1g, s2g);
  tgat_att_kernel<<<dim3(160), dim3(256), 0, stream>>>(
      adj, whT, s1g, s2g, out);
}
